// InstrumentedRIN_75892072121007
// MI455X (gfx1250) — matrix operand order repaired and checked by reading, not compiled, not run on hardware
//
#include <hip/hip_runtime.h>
#include <hip/hip_bf16.h>
#include <math.h>

typedef __attribute__((ext_vector_type(16))) __bf16 v16bf;
typedef __attribute__((ext_vector_type(8)))  __bf16 v8bf;
typedef __attribute__((ext_vector_type(8)))  float  v8f;

#define S_LEN 2048
#define RING  512            // ring depth (power of 2)
#define KTS   1040           // bytes per ktile in LDS A-fragment region (conflict-free stride)

union V16U { v16bf v; v8bf h[2]; };

// ---------------- helpers ----------------

__device__ __forceinline__ float waveReduceSum(float v) {
  #pragma unroll
  for (int off = 16; off > 0; off >>= 1) v += __shfl_xor(v, off, 32);
  return v;
}

__device__ __forceinline__ v8f wmma_bf16(v16bf a, v16bf b, v8f c) {
  return __builtin_amdgcn_wmma_f32_16x16x32_bf16(false, a, false, b, (short)0, c,
                                                 false, false);
}

// A-fragment read: lane reads its 32 bytes (16 bf16) for ktile kt. Two 16B DS loads.
__device__ __forceinline__ v16bf loadA(const unsigned char* afr, int kt, int lane) {
  V16U u;
  u.h[0] = *(const v8bf*)(afr + kt * KTS + lane * 32);
  u.h[1] = *(const v8bf*)(afr + kt * KTS + lane * 32 + 16);
  return u.v;
}

// A-fragment write: thread owns row m (= wave) and 16 consecutive k at k0 = lane*16
// inside one 512-wide K panel -> exactly two contiguous 16B chunks.
__device__ __forceinline__ void storeA(unsigned char* afr, int ktBase, int m, int lane,
                                       const float* x) {
  const int kt  = ktBase + (lane >> 1);
  const int row = m + (lane & 1) * 16; // lane slot: k 0..15 -> m, k 16..31 -> m+16
  v8bf c0, c1;
  #pragma unroll
  for (int j = 0; j < 8; ++j) { c0[j] = (__bf16)x[j]; c1[j] = (__bf16)x[j + 8]; }
  *(v8bf*)(afr + kt * KTS + row * 32)      = c0;
  *(v8bf*)(afr + kt * KTS + row * 32 + 16) = c1;
}

__device__ __forceinline__ void waitCounter(unsigned* p, unsigned target) {
  if (threadIdx.x == 0) {
    while (__hip_atomic_load(p, __ATOMIC_RELAXED, __HIP_MEMORY_SCOPE_AGENT) < target)
      __builtin_amdgcn_s_sleep(1);
  }
  __syncthreads();
  __threadfence();
}

__device__ __forceinline__ void waitProgressAll(unsigned* prog, int n, unsigned target) {
  if (threadIdx.x == 0) {
    for (int i = 0; i < n; ++i)
      while (__hip_atomic_load(&prog[i], __ATOMIC_RELAXED, __HIP_MEMORY_SCOPE_AGENT) < target)
        __builtin_amdgcn_s_sleep(1);
  }
  __syncthreads();
}

__device__ __forceinline__ void releaseAdd(unsigned* p) {
  __threadfence();
  __syncthreads();
  if (threadIdx.x == 0)
    __hip_atomic_fetch_add(p, 1u, __ATOMIC_RELEASE, __HIP_MEMORY_SCOPE_AGENT);
}

// ---------------- prep: pack weights to bf16 B-fragments, zero flags ----------------
__global__ void rin_prep(const float* __restrict__ ipWr, const float* __restrict__ ipWi,
                         const float* __restrict__ ipbr, const float* __restrict__ ipbi,
                         const float* __restrict__ Wqkv, const float* __restrict__ opWr,
                         const float* __restrict__ opWi,
                         __bf16* __restrict__ wc, __bf16* __restrict__ wq,
                         __bf16* __restrict__ wo, float* __restrict__ bsum,
                         unsigned* __restrict__ cnts) {
  const long idx0   = (long)blockIdx.x * blockDim.x + threadIdx.x;
  const long stride = (long)gridDim.x * blockDim.x;

  for (long i = idx0; i < 4112; i += stride) cnts[i] = 0u;
  for (long i = idx0; i < 512; i += stride) bsum[i] = ipbr[i] + ipbi[i];

  // WC: fused complex-linear weights, N=512 (32 nt), K=1024 (32 kt)
  const long NC = 32L * 32 * 512;
  for (long p = idx0; p < NC; p += stride) {
    long nt = p / (32 * 512); long rem = p % (32 * 512);
    long kt = rem / 512;      long q   = rem % 512;
    int lane = (int)(q / 16), j = (int)(q % 16);
    int n = (int)(nt * 16 + (lane & 15));
    int k = (int)(kt * 32 + ((lane < 16) ? j : 16 + j));
    float v = (k < 512) ? (ipWr[n * 512 + k] + ipWi[n * 512 + k])
                        : (ipWr[n * 512 + (k - 512)] - ipWi[n * 512 + (k - 512)]);
    wc[p] = (__bf16)v;
  }
  // WQKV: N=1536 (96 nt), K=512 (16 kt)
  const long NQ = 96L * 16 * 512;
  for (long p = idx0; p < NQ; p += stride) {
    long nt = p / (16 * 512); long rem = p % (16 * 512);
    long kt = rem / 512;      long q   = rem % 512;
    int lane = (int)(q / 16), j = (int)(q % 16);
    int n = (int)(nt * 16 + (lane & 15));
    int k = (int)(kt * 32 + ((lane < 16) ? j : 16 + j));
    wq[p] = (__bf16)Wqkv[n * 512 + k];
  }
  // WOUT: N=1024 (64 nt: cols 0..511 -> opWr, else opWi), K=512
  const long NO = 64L * 16 * 512;
  for (long p = idx0; p < NO; p += stride) {
    long nt = p / (16 * 512); long rem = p % (16 * 512);
    long kt = rem / 512;      long q   = rem % 512;
    int lane = (int)(q / 16), j = (int)(q % 16);
    int n = (int)(nt * 16 + (lane & 15));
    int k = (int)(kt * 32 + ((lane < 16) ? j : 16 + j));
    float v = (n < 512) ? opWr[n * 512 + k] : opWi[(n - 512) * 512 + k];
    wo[p] = (__bf16)v;
  }
}

// ---------------- persistent 3-stage pipeline ----------------
// blocks 0..7  : complex rotation + fused complex-linear  -> x_pre ring
// blocks 8..15 : per-head layernorm + QKV + gated retention -> retr ring
// blocks 16..23: output GEMMs (ore | oim) -> d_out
__global__ void __launch_bounds__(256, 1)
rin_pipeline(const float* __restrict__ w, const float* __restrict__ bg,
             const float* __restrict__ ln_g, const float* __restrict__ ln_b,
             const float* __restrict__ log_alpha,
             const __bf16* __restrict__ wc, const __bf16* __restrict__ wq,
             const __bf16* __restrict__ wo, const float* __restrict__ bsum,
             unsigned* __restrict__ cnt1, unsigned* __restrict__ cnt2,
             unsigned* __restrict__ progA, unsigned* __restrict__ progB,
             float* __restrict__ xpre, float* __restrict__ retr,
             float* __restrict__ out) {
  const int tid  = threadIdx.x;
  const int wave = tid >> 5;
  const int lane = tid & 31;
  const int wg   = blockIdx.x;
  const int m    = wave;          // data row owned by this wave (B = 8 rows, 8 waves)
  const int k0   = lane * 16;     // 16 consecutive K elements per thread

  __shared__ __align__(1024) unsigned char smem[37376];

  if (wg < 8) {
    // -------- stage 1: sync-free producer (carry replicated, kept in registers) ---
    unsigned char* afr  = smem;                 // 32 ktiles * KTS = 33280 B
    unsigned char* cbuf = smem + 33280;         // 4 ntiles * 1024 B partial-C buffer
    for (int i = tid; i < 33280 / 4; i += 256) ((unsigned*)smem)[i] = 0u;
    float hr2[16], hi2[16];
    #pragma unroll
    for (int i = 0; i < 16; ++i) { hr2[i] = 0.0f; hi2[i] = 0.0f; }
    __syncthreads();

    const int ntl = wave & 3;                   // local ntile
    const int nt  = wg * 4 + ntl;               // global ntile in [0,32)
    const int ktb = (wave < 4) ? 0 : 16;        // K split across wave pairs

    for (int t = 0; t < S_LEN; ++t) {
      const int   slot = t & (RING - 1);
      const float tphi = fmodf((float)t * 1.61803398874989f, 6.28318530717959f);
      const float4* wp = (const float4*)(w  + ((long)(m * S_LEN + t) << 9) + k0);
      const float4* bp = (const float4*)(bg + ((long)(m * S_LEN + t) << 9) + k0);
      #pragma unroll
      for (int i = 0; i < 4; ++i) {
        float4 w4 = wp[i], b4 = bp[i];
        const float wv[4] = {w4.x, w4.y, w4.z, w4.w};
        const float bv[4] = {b4.x, b4.y, b4.z, b4.w};
        #pragma unroll
        for (int c = 0; c < 4; ++c) {
          int e = i * 4 + c;
          float rr  = 1.0f / (1.0f + fabsf(wv[c]));
          float thr = hr2[e] * rr + bv[c] + tphi;
          float thi = hi2[e] * rr + bv[c] + tphi;
          float sr = __sinf(thr), cr = __cosf(thr);
          float si = __sinf(thi), ci = __cosf(thi);
          hr2[e] = cr * ci - sr * si;
          hi2[e] = cr * si + sr * ci;
        }
      }
      storeA(afr, 0,  m, lane, hr2);   // K panel [0,512)
      storeA(afr, 16, m, lane, hi2);   // K panel [512,1024)
      __syncthreads();
      if (t >= RING) waitProgressAll(progA, 8, (unsigned)(t - RING + 1));

      // all 8 waves run WMMA: wave pair (w, w+4) splits K of the same ntile.
      // Ktiles processed in pairs: both A fragments (and B tiles) are issued
      // before the first WMMA so DS latency overlaps with matrix math.
      v8f c = {};
      #pragma unroll
      for (int kk = 0; kk < 16; kk += 2) {
        v16bf aA = loadA(afr, ktb + kk,     lane);
        v16bf aB = loadA(afr, ktb + kk + 1, lane);
        v16bf b0 = *((const v16bf*)wc + ((nt * 32 + ktb + kk)     * 32 + lane));
        v16bf b1 = *((const v16bf*)wc + ((nt * 32 + ktb + kk + 1) * 32 + lane));
        c = wmma_bf16(aA, b0, c);
        c = wmma_bf16(aB, b1, c);
      }
      if (wave >= 4) *(v8f*)(cbuf + ntl * 1024 + lane * 32) = c;
      __syncthreads();
      if (wave < 4) {
        v8f cp = *(const v8f*)(cbuf + ntl * 1024 + lane * 32);
        c = c + cp;
        if (lane < 16) {
          int n = nt * 16 + lane;
          float bs = bsum[n];
          float* dst = xpre + ((long)slot << 12) + n;
          #pragma unroll
          for (int r = 0; r < 8; ++r) dst[r << 9] = c[r] + bs;
        }
      }
      releaseAdd(&cnt1[t]);
    }
  } else if (wg < 16) {
    // -------- stage 2: per-head LN + QKV + gated retention --------
    const int h = wg - 8;
    unsigned char* afr = smem;                       // 16 ktiles * KTS = 16640 B
    float* qb   = (float*)(smem + 16640);            // 8 x 64
    float* kb_  = (float*)(smem + 18688);
    float* vb   = (float*)(smem + 20736);
    float* vs   = (float*)(smem + 22784);
    float* gate = (float*)(smem + 24832);
    for (int i = tid; i < 16640 / 4; i += 256) ((unsigned*)smem)[i] = 0u;
    for (int i = tid; i < 512; i += 256) vs[i] = 0.0f;
    __syncthreads();
    const float alpha = 1.0f / (1.0f + __expf(-log_alpha[h]));

    for (int t = 0; t < S_LEN; ++t) {
      const int slot = t & (RING - 1);
      waitCounter(&cnt1[t], 8u);
      // load this wave's row of x_pre straight into registers
      float x[16];
      {
        const float4* sp = (const float4*)(xpre + ((long)slot << 12) + (m << 9) + k0);
        #pragma unroll
        for (int i = 0; i < 4; ++i) {
          float4 v4 = sp[i];
          x[i * 4 + 0] = v4.x; x[i * 4 + 1] = v4.y; x[i * 4 + 2] = v4.z; x[i * 4 + 3] = v4.w;
        }
      }
      // layernorm stats fully in registers (butterfly gives sum to every lane)
      float s = 0.0f, s2 = 0.0f;
      #pragma unroll
      for (int i = 0; i < 16; ++i) { s += x[i]; s2 += x[i] * x[i]; }
      s = waveReduceSum(s); s2 = waveReduceSum(s2);
      const float mean = s * (1.0f / 512.0f);
      const float rstd = rsqrtf(s2 * (1.0f / 512.0f) - mean * mean + 1e-5f);
      {
        const float4* gp = (const float4*)(ln_g + k0);
        const float4* bp = (const float4*)(ln_b + k0);
        #pragma unroll
        for (int i = 0; i < 4; ++i) {
          float4 g4 = gp[i], b4 = bp[i];
          const float gg[4] = {g4.x, g4.y, g4.z, g4.w};
          const float bbv[4] = {b4.x, b4.y, b4.z, b4.w};
          #pragma unroll
          for (int c = 0; c < 4; ++c) {
            int e = i * 4 + c;
            x[e] = (x[e] - mean) * rstd * gg[c] + bbv[c];
          }
        }
      }
      storeA(afr, 0, m, lane, x);
      __syncthreads();
      if (tid == 0)
        __hip_atomic_store(&progA[h], (unsigned)(t + 1), __ATOMIC_RELEASE,
                           __HIP_MEMORY_SCOPE_AGENT);
      // QKV: waves 0-3 -> q & v tiles, waves 4-7 -> k tiles
      if (wave < 4) {
        const int ntq = h * 4 + wave;
        const int ntv = 64 + h * 4 + wave;
        v8f cq = {}, cv = {};
        #pragma unroll
        for (int kt = 0; kt < 16; kt += 2) {
          v16bf aA = loadA(afr, kt,     lane);
          v16bf aB = loadA(afr, kt + 1, lane);
          v16bf bq0 = *((const v16bf*)wq + ((ntq * 16 + kt)     * 32 + lane));
          v16bf bv0 = *((const v16bf*)wq + ((ntv * 16 + kt)     * 32 + lane));
          v16bf bq1 = *((const v16bf*)wq + ((ntq * 16 + kt + 1) * 32 + lane));
          v16bf bv1 = *((const v16bf*)wq + ((ntv * 16 + kt + 1) * 32 + lane));
          cq = wmma_bf16(aA, bq0, cq);
          cv = wmma_bf16(aA, bv0, cv);
          cq = wmma_bf16(aB, bq1, cq);
          cv = wmma_bf16(aB, bv1, cv);
        }
        if (lane < 16) {
          int c0 = wave * 16 + lane;
          #pragma unroll
          for (int r = 0; r < 8; ++r) { qb[r * 64 + c0] = cq[r]; vb[r * 64 + c0] = cv[r]; }
        }
      } else {
        const int ntk = 32 + h * 4 + (wave - 4);
        v8f ck = {};
        #pragma unroll
        for (int kt = 0; kt < 16; kt += 2) {
          v16bf aA = loadA(afr, kt,     lane);
          v16bf aB = loadA(afr, kt + 1, lane);
          v16bf b0 = *((const v16bf*)wq + ((ntk * 16 + kt)     * 32 + lane));
          v16bf b1 = *((const v16bf*)wq + ((ntk * 16 + kt + 1) * 32 + lane));
          ck = wmma_bf16(aA, b0, ck);
          ck = wmma_bf16(aB, b1, ck);
        }
        if (lane < 16) {
          int c0 = (wave - 4) * 16 + lane;
          #pragma unroll
          for (int r = 0; r < 8; ++r) kb_[r * 64 + c0] = ck[r];
        }
      }
      __syncthreads();
      // gate[m] = sigmoid((q.k)/8), wave m handles row m
      {
        float gs = qb[m * 64 + lane] * kb_[m * 64 + lane]
                 + qb[m * 64 + lane + 32] * kb_[m * 64 + lane + 32];
        gs = waveReduceSum(gs);
        if (lane == 0) gate[m] = 1.0f / (1.0f + __expf(-gs * 0.125f));
      }
      __syncthreads();
      if (t >= RING) waitProgressAll(progB, 8, (unsigned)(t - RING + 1));
      #pragma unroll
      for (int i = 0; i < 2; ++i) {
        int e = tid + i * 256;
        int mm = e >> 6, dh = e & 63;
        float vn = alpha * vb[e] + (1.0f - alpha) * vs[e];
        vs[e] = vn;
        retr[((long)slot << 12) + (mm << 9) + h * 64 + dh] = gate[mm] * vn;
      }
      releaseAdd(&cnt2[t]);
    }
  } else {
    // -------- stage 3: output GEMMs, stream ore/oim to d_out --------
    const int jo = wg - 16;
    unsigned char* afr = smem;                       // 16 ktiles * KTS
    for (int i = tid; i < 16640 / 4; i += 256) ((unsigned*)smem)[i] = 0u;
    __syncthreads();
    for (int t = 0; t < S_LEN; ++t) {
      const int slot = t & (RING - 1);
      waitCounter(&cnt2[t], 8u);
      float x[16];
      {
        const float4* sp = (const float4*)(retr + ((long)slot << 12) + (m << 9) + k0);
        #pragma unroll
        for (int i = 0; i < 4; ++i) {
          float4 v4 = sp[i];
          x[i * 4 + 0] = v4.x; x[i * 4 + 1] = v4.y; x[i * 4 + 2] = v4.z; x[i * 4 + 3] = v4.w;
        }
      }
      storeA(afr, 0, m, lane, x);
      __syncthreads();
      if (tid == 0)
        __hip_atomic_store(&progB[jo], (unsigned)(t + 1), __ATOMIC_RELEASE,
                           __HIP_MEMORY_SCOPE_AGENT);
      const int ntg = jo * 8 + wave;                 // global ntile in [0,64)
      v8f c = {};
      #pragma unroll
      for (int kt = 0; kt < 16; kt += 2) {
        v16bf aA = loadA(afr, kt,     lane);
        v16bf aB = loadA(afr, kt + 1, lane);
        v16bf b0 = *((const v16bf*)wo + ((ntg * 16 + kt)     * 32 + lane));
        v16bf b1 = *((const v16bf*)wo + ((ntg * 16 + kt + 1) * 32 + lane));
        c = wmma_bf16(aA, b0, c);
        c = wmma_bf16(aB, b1, c);
      }
      if (lane < 16) {
        int n   = ntg * 16 + lane;
        int cpl = (n >= 512) ? 1 : 0;
        int nn  = n & 511;
        #pragma unroll
        for (int r = 0; r < 8; ++r)
          out[(((long)(cpl * 8 + r) * S_LEN + t) << 9) + nn] = c[r];
      }
      __syncthreads();   // protect afr before next iteration's overwrite
    }
  }
}

// ---------------- host ----------------
extern "C" void kernel_launch(void* const* d_in, const int* in_sizes, int n_in,
                              void* d_out, int out_size, void* d_ws, size_t ws_size,
                              hipStream_t stream) {
  const float* w    = (const float*)d_in[0];
  const float* bg   = (const float*)d_in[1];
  const float* ipWr = (const float*)d_in[2];
  const float* ipWi = (const float*)d_in[3];
  const float* ipbr = (const float*)d_in[4];
  const float* ipbi = (const float*)d_in[5];
  const float* lng  = (const float*)d_in[6];
  const float* lnb  = (const float*)d_in[7];
  const float* Wqkv = (const float*)d_in[8];
  const float* lal  = (const float*)d_in[9];
  const float* opWr = (const float*)d_in[10];
  const float* opWi = (const float*)d_in[11];
  float* out = (float*)d_out;

  auto align256 = [](size_t x) { return (x + 255) & ~(size_t)255; };
  size_t off = 0;
  size_t o_wc   = off; off = align256(off + 32L * 32 * 512 * 2);
  size_t o_wq   = off; off = align256(off + 96L * 16 * 512 * 2);
  size_t o_wo   = off; off = align256(off + 64L * 16 * 512 * 2);
  size_t o_bsum = off; off = align256(off + 512 * 4);
  size_t o_cnt  = off; off = align256(off + 4112 * 4);
  size_t o_xpre = off; off = align256(off + (size_t)RING * 4096 * 4);
  size_t o_retr = off; off = align256(off + (size_t)RING * 4096 * 4);
  if (ws_size < off) return;

  char* ws = (char*)d_ws;
  __bf16*   wc    = (__bf16*)(ws + o_wc);
  __bf16*   wq    = (__bf16*)(ws + o_wq);
  __bf16*   wo    = (__bf16*)(ws + o_wo);
  float*    bsum  = (float*)(ws + o_bsum);
  unsigned* cnts  = (unsigned*)(ws + o_cnt);
  unsigned* cnt1  = cnts;
  unsigned* cnt2  = cnts + 2048;
  unsigned* progA = cnts + 4096;
  unsigned* progB = cnts + 4104;
  float*    xpre  = (float*)(ws + o_xpre);
  float*    retr  = (float*)(ws + o_retr);

  rin_prep<<<256, 256, 0, stream>>>(ipWr, ipWi, ipbr, ipbi, Wqkv, opWr, opWi,
                                    wc, wq, wo, bsum, cnts);
  rin_pipeline<<<24, 256, 0, stream>>>(w, bg, lng, lnb, lal, wc, wq, wo, bsum,
                                       cnt1, cnt2, progA, progB, xpre, retr, out);
}
